// EncoderMultiHeadAttention_2671469658563
// MI455X (gfx1250) — matrix operand order repaired and checked by reading, not compiled, not run on hardware
//
#include <hip/hip_runtime.h>
#include <hip/hip_bf16.h>
#include <math.h>

// ---------------------------------------------------------------------------
// EncoderMultiHeadAttention on gfx1250 (CDNA5, wave32, WMMA)
//   B=2, L=2048, D=1024, H=16, DK=64
// Pipeline:
//   cvt weights->bf16 | LN->bf16 | QKV WMMA GEMMs (V stored transposed) |
//   FUSED flash-style scores+softmax+attn@V (single attn write) |
//   out-proj WMMA + residual
// ---------------------------------------------------------------------------

typedef __bf16 bf16;
typedef __attribute__((ext_vector_type(16))) __bf16 v16bf;
typedef __attribute__((ext_vector_type(8)))  __bf16 v8bf;
typedef __attribute__((ext_vector_type(4)))  __bf16 v4bf;
typedef __attribute__((ext_vector_type(8)))  float  v8f;

#define B_  2
#define L_  2048
#define D_  1024
#define H_  16
#define DK_ 64
#define M_  (B_ * L_)      // 4096 rows
#define EPS_ 1e-5f

union FragBF { v16bf v; v8bf h[2]; };

__device__ __forceinline__ v8f zero_v8f() {
    v8f a;
#pragma unroll
    for (int j = 0; j < 8; ++j) a[j] = 0.0f;
    return a;
}

// A fragment, 16x32 bf16, row-major source [.. x ld]:
// lanes 0-15: row m, K { kb+0..7, kb+16..23 }; lanes 16-31: K { kb+8..15, kb+24..31 }
__device__ __forceinline__ v16bf load_fragA_bf16(const bf16* __restrict__ base,
                                                 int ld, int row, int kb, int lane) {
    FragBF f;
    const int klo = (lane & 16) ? 8 : 0;
    const bf16* p = base + (size_t)row * ld + kb + klo;
    f.h[0] = *(const v8bf*)(p);
    f.h[1] = *(const v8bf*)(p + 16);
    return f.v;
}

// B fragment, 32x16 bf16, B[k][n] supplied as row-major [N x K] (row n = column n):
// lanes 0-15: col n, K { kb+0..7, kb+16..23 }; lanes 16-31: K { kb+8..15, kb+24..31 }
__device__ __forceinline__ v16bf load_fragB_bf16(const bf16* __restrict__ base,
                                                 int ld, int col, int kb, int lane) {
    FragBF f;
    const int ko = (lane & 16) ? 8 : 0;
    const bf16* p = base + (size_t)col * ld + kb + ko;
    f.h[0] = *(const v8bf*)(p);
    f.h[1] = *(const v8bf*)(p + 16);
    return f.v;
}

__device__ __forceinline__ v8f wmma_bf16(v16bf a, v16bf b, v8f c) {
    return __builtin_amdgcn_wmma_f32_16x16x32_bf16(false, a, false, b,
                                                   (short)0, c, false, false);
}

// ---------------------------------------------------------------------------
// Kernel 1: f32 -> bf16 weight conversion (vectorized 4/thread)
// ---------------------------------------------------------------------------
__global__ void cvt_bf16_kernel(const float* __restrict__ src, bf16* __restrict__ dst, int n4) {
    int i = blockIdx.x * blockDim.x + threadIdx.x;
    if (i < n4) {
        float4 f = ((const float4*)src)[i];
        v4bf o;
        o[0] = (bf16)f.x; o[1] = (bf16)f.y; o[2] = (bf16)f.z; o[3] = (bf16)f.w;
        ((v4bf*)dst)[i] = o;
    }
}

// ---------------------------------------------------------------------------
// Kernel 2: row LayerNorm (D=1024), one 256-thread block per row, bf16 out
// ---------------------------------------------------------------------------
__global__ void ln_bf16_kernel(const float* __restrict__ x, const float* __restrict__ g,
                               const float* __restrict__ bb, bf16* __restrict__ y) {
    __shared__ float sred[2][8];
    const int row = blockIdx.x;
    const int t = threadIdx.x;
    const float* xr = x + (size_t)row * D_;
    float4 f = ((const float4*)xr)[t];
    float s  = f.x + f.y + f.z + f.w;
    float ss = f.x * f.x + f.y * f.y + f.z * f.z + f.w * f.w;
#pragma unroll
    for (int o = 16; o; o >>= 1) {           // wave32 reduction
        s  += __shfl_xor(s,  o, 32);
        ss += __shfl_xor(ss, o, 32);
    }
    const int wv = t >> 5, lane = t & 31;
    if (lane == 0) { sred[0][wv] = s; sred[1][wv] = ss; }
    __syncthreads();
    float S = 0.f, SS = 0.f;
#pragma unroll
    for (int w = 0; w < 8; ++w) { S += sred[0][w]; SS += sred[1][w]; }
    const float mu  = S * (1.0f / D_);
    const float var = SS * (1.0f / D_) - mu * mu;
    const float rs  = rsqrtf(var + EPS_);
    float4 gg = ((const float4*)g)[t];
    float4 b4 = ((const float4*)bb)[t];
    v4bf o;
    o[0] = (bf16)((f.x - mu) * rs * gg.x + b4.x);
    o[1] = (bf16)((f.y - mu) * rs * gg.y + b4.y);
    o[2] = (bf16)((f.z - mu) * rs * gg.z + b4.z);
    o[3] = (bf16)((f.w - mu) * rs * gg.w + b4.w);
    ((v4bf*)(y + (size_t)row * D_))[t] = o;
}

// ---------------------------------------------------------------------------
// Kernel 3: Y = X[M x K] @ W[N x K]^T  via v_wmma_f32_16x16x32_bf16
//   one 16x16 output tile per wave; 8 waves / block
//   mode 0: Y -> bf16 head-major  [B,H,L,DK]
//   mode 1: Y -> bf16 head-transposed [B,H,DK,L]
//   mode 2: Y -> f32 [M,D] = acc + residual   (final projection)
// ---------------------------------------------------------------------------
__global__ void gemm_xw_kernel(const bf16* __restrict__ X, const bf16* __restrict__ W,
                               bf16* __restrict__ Ybf, float* __restrict__ Yf32,
                               const float* __restrict__ resid, int mode) {
    const int t = threadIdx.x;
    const int lane = t & 31, wv = t >> 5;
    const size_t tile = (size_t)blockIdx.x * 8 + wv;
    const int tilesN = D_ / 16;                    // 64
    const int tm = (int)(tile / tilesN);
    const int tn = (int)(tile % tilesN);
    const int m0 = tm * 16, n0 = tn * 16;
    const int rowA = m0 + (lane & 15);
    const int rowB = n0 + (lane & 15);

    v8f acc = zero_v8f();
    for (int kb = 0; kb < D_; kb += 32) {
        v16bf a = load_fragA_bf16(X, D_, rowA, kb, lane);
        v16bf b = load_fragB_bf16(W, D_, rowB, kb, lane);
        acc = wmma_bf16(a, b, acc);
    }

    const int col   = n0 + (lane & 15);
    const int rbase = m0 + ((lane & 16) ? 8 : 0);
#pragma unroll
    for (int j = 0; j < 8; ++j) {
        const int r = rbase + j;
        if (mode == 2) {
            const size_t idx = (size_t)r * D_ + col;
            Yf32[idx] = acc[j] + resid[idx];
        } else {
            const int bI = r / L_, l = r % L_;
            const int h = col / DK_, dk = col % DK_;
            const size_t idx = (mode == 0)
                ? ((((size_t)bI * H_ + h) * L_ + l) * DK_ + dk)
                : ((((size_t)bI * H_ + h) * DK_ + dk) * L_ + l);
            Ybf[idx] = (bf16)acc[j];
        }
    }
}

// ---------------------------------------------------------------------------
// Kernel 4 (FUSED): scores + softmax + attn@V for 16 query rows of one head.
//   128 threads = 4 waves; wave w owns key tiles [w*32, w*32+32).
//   Pass 1: online row max/sum (16-lane shuffles, WMMA scores)
//   Pass 2: recompute scores, write final attn (f32, once), stage bf16 tile
//           in wave-private LDS, accumulate attn@V with WMMA against V^T.
//   Pass 3: cross-wave partial-O reduction in LDS.
// ---------------------------------------------------------------------------
__global__ void __launch_bounds__(128)
attn_fused_kernel(const bf16* __restrict__ qh, const bf16* __restrict__ kh,
                  const bf16* __restrict__ vhT, const int* __restrict__ mask,
                  float* __restrict__ attnOut, bf16* __restrict__ Obuf) {
    __shared__ float smax[4][16];
    __shared__ float ssum[4][16];
    __shared__ __align__(16) bf16 aT[4][16 * 32];     // per-wave attn staging tile
    __shared__ float oRed[4][4][8][32];               // [wave][ntile][e][lane]

    const int t = threadIdx.x;
    const int lane = t & 31, w = t >> 5;
    const int bh = blockIdx.x >> 7;                   // / (L_/16)
    const int ti = blockIdx.x & 127;
    const int i0 = ti * 16;
    const int bI = bh / H_, h = bh % H_;

    const bf16* qp = qh  + (size_t)bh * L_ * DK_;
    const bf16* kp = kh  + (size_t)bh * L_ * DK_;
    const bf16* vp = vhT + (size_t)bh * DK_ * L_;
    const int*  mp = mask + (size_t)bI * L_ * L_;
    float*      ap = attnOut + (size_t)bh * L_ * L_;

    const int rowA  = i0 + (lane & 15);
    const int lrow  = (lane & 16) ? 8 : 0;            // local row base
    const int ibase = i0 + lrow;                      // global query row base

    // Q fragments for K = 0..63, loaded once
    const v16bf aQ0 = load_fragA_bf16(qp, DK_, rowA, 0,  lane);
    const v16bf aQ1 = load_fragA_bf16(qp, DK_, rowA, 32, lane);

    float rm[8], rs[8];
#pragma unroll
    for (int e = 0; e < 8; ++e) { rm[e] = -3.0e38f; rs[e] = 0.f; }

    const int jt0 = w * 32;

    // ---- pass 1: online max/sum over this wave's 32 key tiles ----
    for (int jt = jt0; jt < jt0 + 32; ++jt) {
        const int j0   = jt * 16;
        const int colB = j0 + (lane & 15);
        v8f s = zero_v8f();
        s = wmma_bf16(aQ0, load_fragB_bf16(kp, DK_, colB, 0,  lane), s);
        s = wmma_bf16(aQ1, load_fragB_bf16(kp, DK_, colB, 32, lane), s);
        // warm V^T tile for pass 2 (global_prefetch_b8)
        __builtin_prefetch(vp + (size_t)(lane & 15) * L_ + j0, 0, 1);
#pragma unroll
        for (int e = 0; e < 8; ++e) {
            const float sv = (mp[(size_t)(ibase + e) * L_ + colB] == 0)
                                 ? -__builtin_inff() : s[e] * 0.125f;
            float tm = sv;
#pragma unroll
            for (int o = 1; o < 16; o <<= 1) tm = fmaxf(tm, __shfl_xor(tm, o, 32));
            const float nm = fmaxf(rm[e], tm);
            float p = __expf(sv - nm);
            float ts = p;
#pragma unroll
            for (int o = 1; o < 16; o <<= 1) ts += __shfl_xor(ts, o, 32);
            rs[e] = rs[e] * __expf(rm[e] - nm) + ts;
            rm[e] = nm;
        }
    }

    // publish per-wave stats, combine across waves
    if ((lane & 15) == 0) {
#pragma unroll
        for (int e = 0; e < 8; ++e) { smax[w][lrow + e] = rm[e]; ssum[w][lrow + e] = rs[e]; }
    }
    __syncthreads();
    float Mrow[8], Sinv[8];
#pragma unroll
    for (int e = 0; e < 8; ++e) {
        const int r = lrow + e;
        float mm = smax[0][r];
#pragma unroll
        for (int w2 = 1; w2 < 4; ++w2) mm = fmaxf(mm, smax[w2][r]);
        float s2 = 0.f;
#pragma unroll
        for (int w2 = 0; w2 < 4; ++w2) s2 += ssum[w2][r] * __expf(smax[w2][r] - mm);
        Mrow[e] = mm;
        Sinv[e] = 1.0f / s2;
    }

    // ---- pass 2: recompute, emit attn once, accumulate attn@V ----
    v8f accO[4];
#pragma unroll
    for (int n = 0; n < 4; ++n) accO[n] = zero_v8f();
    bf16* myT = &aT[w][0];

    for (int pr = 0; pr < 16; ++pr) {
        const int jbase = (jt0 + pr * 2) * 16;        // K offset (32 keys)
#pragma unroll
        for (int sub = 0; sub < 2; ++sub) {
            const int j0   = jbase + sub * 16;
            const int colB = j0 + (lane & 15);
            v8f s = zero_v8f();
            s = wmma_bf16(aQ0, load_fragB_bf16(kp, DK_, colB, 0,  lane), s);
            s = wmma_bf16(aQ1, load_fragB_bf16(kp, DK_, colB, 32, lane), s);
#pragma unroll
            for (int e = 0; e < 8; ++e) {
                const float sv = (mp[(size_t)(ibase + e) * L_ + colB] == 0)
                                     ? -__builtin_inff() : s[e] * 0.125f;
                const float a = __expf(sv - Mrow[e]) * Sinv[e];
                ap[(size_t)(ibase + e) * L_ + colB] = a;                      // final attn
                myT[(lrow + e) * 32 + sub * 16 + (lane & 15)] = (bf16)a;      // stage
            }
        }
        // wave-private LDS, in-order per wave: safe to read back without barrier
        const v16bf aF = load_fragA_bf16(myT, 32, lane & 15, 0, lane);
#pragma unroll
        for (int n = 0; n < 4; ++n) {
            const v16bf bF = load_fragB_bf16(vp, L_, n * 16 + (lane & 15), jbase, lane);
            accO[n] = wmma_bf16(aF, bF, accO[n]);
        }
    }

    // ---- pass 3: cross-wave reduction of partial O ----
#pragma unroll
    for (int n = 0; n < 4; ++n)
#pragma unroll
        for (int e = 0; e < 8; ++e) oRed[w][n][e][lane] = accO[n][e];
    __syncthreads();
    {
        const int n = w;                              // wave w writes d-tile w
        const int dcol = n * 16 + (lane & 15);
#pragma unroll
        for (int e = 0; e < 8; ++e) {
            const float sum = oRed[0][n][e][lane] + oRed[1][n][e][lane]
                            + oRed[2][n][e][lane] + oRed[3][n][e][lane];
            Obuf[((size_t)bI * L_ + ibase + e) * D_ + h * DK_ + dcol] = (bf16)sum;
        }
    }
}

// ---------------------------------------------------------------------------
// Host launcher
// ---------------------------------------------------------------------------
extern "C" void kernel_launch(void* const* d_in, const int* in_sizes, int n_in,
                              void* d_out, int out_size, void* d_ws, size_t ws_size,
                              hipStream_t stream) {
    const float* q    = (const float*)d_in[0];
    const float* k    = (const float*)d_in[1];
    const float* v    = (const float*)d_in[2];
    const int*   mask = (const int*)d_in[3];
    const float* gq = (const float*)d_in[4];
    const float* bq = (const float*)d_in[5];
    const float* gk = (const float*)d_in[6];
    const float* bk = (const float*)d_in[7];
    const float* gv = (const float*)d_in[8];
    const float* bv = (const float*)d_in[9];
    const float* Wq  = (const float*)d_in[10];
    const float* Wk  = (const float*)d_in[11];
    const float* Wv  = (const float*)d_in[12];
    const float* Wfc = (const float*)d_in[13];

    float* outP  = (float*)d_out;                       // [B,L,D] f32
    float* attnP = outP + (size_t)B_ * L_ * D_;         // [B,H,L,L] f32

    char* ws = (char*)d_ws;
    const size_t WBYTES = (size_t)D_ * D_ * sizeof(bf16);   // 2 MB per weight
    const size_t XBYTES = (size_t)M_ * D_ * sizeof(bf16);   // 8 MB per activation
    bf16* wqb = (bf16*)(ws + 0 * WBYTES);
    bf16* wkb = (bf16*)(ws + 1 * WBYTES);
    bf16* wvb = (bf16*)(ws + 2 * WBYTES);
    bf16* wfb = (bf16*)(ws + 3 * WBYTES);
    bf16* xnq = (bf16*)(ws + 4 * WBYTES + 0 * XBYTES);
    bf16* xnk = (bf16*)(ws + 4 * WBYTES + 1 * XBYTES);
    bf16* xnv = (bf16*)(ws + 4 * WBYTES + 2 * XBYTES);
    bf16* qh  = (bf16*)(ws + 4 * WBYTES + 3 * XBYTES);
    bf16* kh  = (bf16*)(ws + 4 * WBYTES + 4 * XBYTES);
    bf16* vhT = (bf16*)(ws + 4 * WBYTES + 5 * XBYTES);
    bf16* ob  = (bf16*)(ws + 4 * WBYTES + 6 * XBYTES);      // total 64 MB

    dim3 blk(256);

    // 1) weights -> bf16
    const int n4 = D_ * D_ / 4;
    cvt_bf16_kernel<<<n4 / 256, blk, 0, stream>>>(Wq,  wqb, n4);
    cvt_bf16_kernel<<<n4 / 256, blk, 0, stream>>>(Wk,  wkb, n4);
    cvt_bf16_kernel<<<n4 / 256, blk, 0, stream>>>(Wv,  wvb, n4);
    cvt_bf16_kernel<<<n4 / 256, blk, 0, stream>>>(Wfc, wfb, n4);

    // 2) LayerNorm -> bf16
    ln_bf16_kernel<<<M_, blk, 0, stream>>>(q, gq, bq, xnq);
    ln_bf16_kernel<<<M_, blk, 0, stream>>>(k, gk, bk, xnk);
    ln_bf16_kernel<<<M_, blk, 0, stream>>>(v, gv, bv, xnv);

    // 3) QKV projections (WMMA): q/k head-major, v head-transposed
    const int gemmBlocks = (M_ / 16) * (D_ / 16) / 8;   // 2048
    gemm_xw_kernel<<<gemmBlocks, blk, 0, stream>>>(xnq, wqb, qh,  nullptr, nullptr, 0);
    gemm_xw_kernel<<<gemmBlocks, blk, 0, stream>>>(xnk, wkb, kh,  nullptr, nullptr, 0);
    gemm_xw_kernel<<<gemmBlocks, blk, 0, stream>>>(xnv, wvb, vhT, nullptr, nullptr, 1);

    // 4) fused scores + softmax + attn@V (single attn materialization)
    attn_fused_kernel<<<B_ * H_ * (L_ / 16), dim3(128), 0, stream>>>(
        qh, kh, vhT, mask, attnP, ob);

    // 5) output projection + residual (WMMA) -> f32 out
    gemm_xw_kernel<<<gemmBlocks, blk, 0, stream>>>(ob, wfb, nullptr, outP, q, 2);
}
